// LinearAttention_53730040873608
// MI455X (gfx1250) — matrix operand order repaired and checked by reading, not compiled, not run on hardware
//
#include <hip/hip_runtime.h>
#include <stdint.h>

// ---------------------------------------------------------------------------
// Problem constants
// ---------------------------------------------------------------------------
#define BQ      4
#define SQ      2048
#define NTOK    8192            // BQ * SQ
#define FEAT    1024
#define INTER   2048
#define THREE_I 6144            // 3 * INTER
#define NEXP    8
#define KW      7
#define GIC     512             // INTER / BOTTLENECK_GROUP
#define GOC     1536            // THREE_I / BOTTLENECK_GROUP

// GEMM tiling (wave32, 8 waves / block)
#define MT   128
#define NT   128
#define KT   64
#define LDSK 72                 // padded K stride in LDS (bf16 elems), 144B rows
#define MEXT 134                // 128 + (KW-1) rows for conv sliding window

typedef __bf16 bhalf_t;
typedef bhalf_t v16bf __attribute__((ext_vector_type(16)));
typedef float   v8f   __attribute__((ext_vector_type(8)));
typedef int     vi4   __attribute__((ext_vector_type(4)));

union FragAB { v16bf v; uint4 q[2]; };

// --- CDNA5 async global->LDS copy (ASYNCcnt) with safe fallback -------------
#if __has_builtin(__builtin_amdgcn_global_load_async_to_lds_b128) && \
    __has_builtin(__builtin_amdgcn_s_wait_asynccnt)
#define HAVE_ASYNC 1
#else
#define HAVE_ASYNC 0
#endif

__device__ __forceinline__ void cp16(const unsigned short* g, unsigned short* l) {
#if HAVE_ASYNC
    __builtin_amdgcn_global_load_async_to_lds_b128(
        (__attribute__((address_space(1))) vi4*)
            (__attribute__((address_space(1))) void*)(void*)(g),
        (__attribute__((address_space(3))) vi4*)
            (__attribute__((address_space(3))) void*)(void*)(l),
        0, 0);
#else
    *(uint4*)l = *(const uint4*)g;
#endif
}

#if HAVE_ASYNC
#define STAGE_WAIT_KEEP8() __builtin_amdgcn_s_wait_asynccnt(8)
#define STAGE_WAIT_ALL()   __builtin_amdgcn_s_wait_asynccnt(0)
#else
#define STAGE_WAIT_KEEP8()
#define STAGE_WAIT_ALL()
#endif

__device__ __forceinline__ unsigned short f2bf(float f) {
    unsigned int u = __float_as_uint(f);
    unsigned int r = u + 0x7FFFu + ((u >> 16) & 1u);   // round-to-nearest-even
    return (unsigned short)(r >> 16);
}

__device__ __forceinline__ v8f wmma_bf16(const FragAB& a, const FragAB& b, v8f c) {
    return __builtin_amdgcn_wmma_f32_16x16x32_bf16(false, a.v, false, b.v,
                                                   (short)0, c, false, false);
}

// A fragment: 16x32 bf16. lane<16: K {0..7,16..23}; lane>=16: K {8..15,24..31}
__device__ __forceinline__ FragAB load_frag_A(const unsigned short* As, int rowBase,
                                              int ks, int lane) {
    int r = lane & 15, half = lane >> 4;
    const unsigned short* p = As + (rowBase + r) * LDSK + ks + half * 8;
    FragAB f;
    f.q[0] = *(const uint4*)p;
    f.q[1] = *(const uint4*)(p + 16);
    return f;
}

// B fragment: 32x16 bf16 from LDS stored [N][K]. lane<16: K {0..7,16..23}; lane>=16: K {8..15,24..31}
__device__ __forceinline__ FragAB load_frag_B(const unsigned short* Bs, int colBase,
                                              int ks, int lane) {
    int n = lane & 15, half = lane >> 4;
    const unsigned short* p = Bs + (colBase + n) * LDSK + ks + half * 8;
    FragAB f;
    f.q[0] = *(const uint4*)p;
    f.q[1] = *(const uint4*)(p + 16);
    return f;
}

// 16 WMMAs on one staged K-tile
__device__ __forceinline__ void gemm_tile_compute(const unsigned short* As,
                                                  const unsigned short* Bs,
                                                  int rowOff, int wm, int wn, int lane,
                                                  v8f (&acc)[4][2]) {
    #pragma unroll
    for (int ks = 0; ks < KT; ks += 32) {
        FragAB af[4], bfr[2];
        #pragma unroll
        for (int i = 0; i < 4; i++) af[i]  = load_frag_A(As, rowOff + wm + i * 16, ks, lane);
        #pragma unroll
        for (int j = 0; j < 2; j++) bfr[j] = load_frag_B(Bs, wn + j * 16, ks, lane);
        #pragma unroll
        for (int i = 0; i < 4; i++)
            #pragma unroll
            for (int j = 0; j < 2; j++)
                acc[i][j] = wmma_bf16(af[i], bfr[j], acc[i][j]);
    }
}

// ---------------------------------------------------------------------------
// 0) inverse feature shuffles (so repacked weights absorb the column shuffle)
// ---------------------------------------------------------------------------
__global__ void inv_perm_kernel(const int* __restrict__ sh0, const int* __restrict__ sh2,
                                int* __restrict__ inv0, int* __restrict__ inv2) {
    int t = blockIdx.x * 256 + threadIdx.x;
    if (t < FEAT)  inv0[sh0[t]] = t;
    if (t < INTER) inv2[sh2[t]] = t;
}

// ---------------------------------------------------------------------------
// 1) transpose inp (B,F,S) fp32 -> token-major bf16 XT[tok][f]
// ---------------------------------------------------------------------------
__global__ __launch_bounds__(256) void transpose_in_kernel(
        const float* __restrict__ Inp, unsigned short* __restrict__ XT) {
    __shared__ float tile[32][33];
    int b = blockIdx.z, s0 = blockIdx.x * 32, f0 = blockIdx.y * 32;
    int tx = threadIdx.x, ty = threadIdx.y;
    #pragma unroll
    for (int yy = 0; yy < 4; yy++) {
        int f = f0 + ty + yy * 8;
        tile[ty + yy * 8][tx] = Inp[((size_t)b * FEAT + f) * SQ + s0 + tx];
    }
    __syncthreads();
    #pragma unroll
    for (int yy = 0; yy < 4; yy++) {
        int s = s0 + ty + yy * 8;
        XT[((size_t)(b * SQ + s)) * FEAT + f0 + tx] = f2bf(tile[tx][ty + yy * 8]);
    }
}

// ---------------------------------------------------------------------------
// 2) weight repacks to bf16, transposed to [N][K], shuffle folded in
// ---------------------------------------------------------------------------
__global__ __launch_bounds__(256) void repack_w0_kernel(
        const float* __restrict__ W0, const int* __restrict__ inv0,
        unsigned short* __restrict__ W0s) {             // [e][c(6144)][f(1024)]
    __shared__ float tile[32][33];
    int e = blockIdx.z, c0 = blockIdx.x * 32, f0 = blockIdx.y * 32;
    int tx = threadIdx.x, ty = threadIdx.y;
    #pragma unroll
    for (int yy = 0; yy < 4; yy++) {
        int fr = inv0[f0 + ty + yy * 8];
        tile[ty + yy * 8][tx] = W0[((size_t)e * FEAT + fr) * THREE_I + c0 + tx];
    }
    __syncthreads();
    #pragma unroll
    for (int yy = 0; yy < 4; yy++) {
        int c = c0 + ty + yy * 8;
        W0s[((size_t)e * THREE_I + c) * FEAT + f0 + tx] = f2bf(tile[tx][ty + yy * 8]);
    }
}

__global__ __launch_bounds__(256) void repack_w2_kernel(
        const float* __restrict__ W2, const int* __restrict__ inv2,
        unsigned short* __restrict__ W2s) {             // [e][c(1024)][f(2048)]
    __shared__ float tile[32][33];
    int e = blockIdx.z, c0 = blockIdx.x * 32, f0 = blockIdx.y * 32;
    int tx = threadIdx.x, ty = threadIdx.y;
    #pragma unroll
    for (int yy = 0; yy < 4; yy++) {
        int fr = inv2[f0 + ty + yy * 8];
        tile[ty + yy * 8][tx] = W2[((size_t)e * INTER + fr) * FEAT + c0 + tx];
    }
    __syncthreads();
    #pragma unroll
    for (int yy = 0; yy < 4; yy++) {
        int c = c0 + ty + yy * 8;
        W2s[((size_t)e * FEAT + c) * INTER + f0 + tx] = f2bf(tile[tx][ty + yy * 8]);
    }
}

__global__ __launch_bounds__(128) void repack_w1_kernel(
        const float* __restrict__ W1, unsigned short* __restrict__ W1s) { // [k][o][i]
    int o = blockIdx.x;
    for (int i = threadIdx.x; i < GIC; i += 128) {
        const float* p = W1 + ((size_t)o * GIC + i) * KW;
        #pragma unroll
        for (int k = 0; k < KW; k++)
            W1s[((size_t)k * THREE_I + o) * GIC + i] = f2bf(p[k]);
    }
}

// ---------------------------------------------------------------------------
// 3) MoE GEMM (moe0 and moe2): double-buffered async LDS pipeline.
//    Out[e + 8*m][n] = sum_k A[perm[e+8*m]][k] * Wt[e][n][k]
// ---------------------------------------------------------------------------
__global__ __launch_bounds__(256) void moe_gemm_kernel(
        const unsigned short* __restrict__ A,    // [NTOK][K] bf16
        const unsigned short* __restrict__ Wt,   // [NEXP][N][K] bf16
        const int* __restrict__ perm,
        float* __restrict__ Out,                 // [NTOK][N] fp32
        int K, int N) {
    __shared__ alignas(16) unsigned short As[2][MT * LDSK];
    __shared__ alignas(16) unsigned short Bs[2][NT * LDSK];
    __shared__ int rowsrc[MT];

    const int tid  = threadIdx.x;
    const int lane = tid & 31;
    const int wave = tid >> 5;
    const int wm   = (wave & 1) * 64;
    const int wn   = (wave >> 1) * 32;

    const int e  = blockIdx.z;
    const int m0 = blockIdx.y * MT;
    const int n0 = blockIdx.x * NT;

    if (tid < MT) rowsrc[tid] = perm[e + NEXP * (m0 + tid)];
    const unsigned short* W = Wt + (size_t)e * N * K;
    __syncthreads();

    v8f acc[4][2];
    #pragma unroll
    for (int i = 0; i < 4; i++)
        #pragma unroll
        for (int j = 0; j < 2; j++)
            #pragma unroll
            for (int v = 0; v < 8; v++) acc[i][j][v] = 0.0f;

    auto issue_stage = [&](int k0, int buf) {
        #pragma unroll
        for (int p = 0; p < 4; p++) {                 // A: 128x64 bf16 (16KB)
            int seg = tid + p * 256;
            int r = seg >> 3, s = seg & 7;
            cp16(A + (size_t)rowsrc[r] * K + k0 + s * 8, &As[buf][r * LDSK + s * 8]);
        }
        #pragma unroll
        for (int p = 0; p < 4; p++) {                 // B: 128x64 bf16 (16KB)
            int seg = tid + p * 256;
            int r = seg >> 3, s = seg & 7;
            cp16(W + (size_t)(n0 + r) * K + k0 + s * 8, &Bs[buf][r * LDSK + s * 8]);
        }
    };

    const int nIter = K / KT;
    issue_stage(0, 0);
    for (int it = 0; it < nIter; ++it) {
        int cur = it & 1;
        if (it + 1 < nIter) {
            issue_stage((it + 1) * KT, cur ^ 1);
            STAGE_WAIT_KEEP8();                       // current stage landed
        } else {
            STAGE_WAIT_ALL();
        }
        __syncthreads();
        gemm_tile_compute(As[cur], Bs[cur], 0, wm, wn, lane, acc);
        __syncthreads();                              // readers done before overwrite
    }

    const int half = lane >> 4, nl = lane & 15;
    #pragma unroll
    for (int i = 0; i < 4; i++)
        #pragma unroll
        for (int j = 0; j < 2; j++)
            #pragma unroll
            for (int v = 0; v < 8; v++) {
                int mrow = m0 + wm + i * 16 + v + 8 * half;
                int jrow = e + NEXP * mrow;
                Out[(size_t)jrow * N + n0 + wn + j * 16 + nl] = acc[i][j][v];
            }
}

// ---------------------------------------------------------------------------
// 4) cumsum gate: U[b*S+s][c] = relu(cumsum(depth)/(s+1))^3 * scale + shift
// ---------------------------------------------------------------------------
__global__ __launch_bounds__(256) void cumsum_gate_kernel(
        const float* __restrict__ Of, const int* __restrict__ perm,
        float* __restrict__ U) {
    __shared__ int pl[SQ];
    int b = blockIdx.y;
    int c = blockIdx.x * 256 + threadIdx.x;
    for (int s = threadIdx.x; s < SQ; s += 256) pl[s] = perm[b * SQ + s];
    __syncthreads();
    float cum = 0.0f;
    for (int s = 0; s < SQ; s++) {
        const float* row = Of + (size_t)pl[s] * THREE_I;
        if (s + 1 < SQ) {                 // prefetch next gathered row segments
            const float* nrow = Of + (size_t)pl[s + 1] * THREE_I;
            __builtin_prefetch(nrow + c, 0, 1);
            __builtin_prefetch(nrow + c + INTER, 0, 1);
            __builtin_prefetch(nrow + c + 2 * INTER, 0, 1);
        }
        float d  = row[c];
        float sc = row[c + INTER];
        float sh = row[c + 2 * INTER];
        cum += d;
        float m = cum / (float)(s + 1);
        float r = m > 0.0f ? m : 0.0f;
        U[(size_t)(b * SQ + s) * INTER + c] = r * r * r * sc + sh;
    }
}

// ---------------------------------------------------------------------------
// channel-norm (center + L2 normalize * sqrt(C)) helpers + kernels
// ---------------------------------------------------------------------------
__device__ __forceinline__ void block_reduce2(float& a, float& b) {
    #pragma unroll
    for (int off = 16; off > 0; off >>= 1) {
        a += __shfl_down(a, off);
        b += __shfl_down(b, off);
    }
    __shared__ float ra[8], rb[8];
    int wave = threadIdx.x >> 5, lane = threadIdx.x & 31;
    if (lane == 0) { ra[wave] = a; rb[wave] = b; }
    __syncthreads();
    float ta = 0.0f, tb = 0.0f;
    #pragma unroll
    for (int w = 0; w < 8; w++) { ta += ra[w]; tb += rb[w]; }
    a = ta; b = tb;
}

__global__ __launch_bounds__(256) void norm1_kernel(
        const float* __restrict__ U, unsigned short* __restrict__ Xn) {
    int i = blockIdx.x;
    const float* row = U + (size_t)i * INTER;
    float vals[8], s1 = 0.0f, s2 = 0.0f;
    #pragma unroll
    for (int q = 0; q < 8; q++) {
        float v = row[threadIdx.x + q * 256];
        vals[q] = v; s1 += v; s2 += v * v;
    }
    block_reduce2(s1, s2);
    float mean  = s1 * (1.0f / INTER);
    float var   = s2 - (float)INTER * mean * mean;
    float scale = sqrtf((float)INTER) * rsqrtf(var);
    #pragma unroll
    for (int q = 0; q < 8; q++)
        Xn[(size_t)i * INTER + threadIdx.x + q * 256] = f2bf((vals[q] - mean) * scale);
}

__global__ __launch_bounds__(256) void norm2_kernel(
        const float* __restrict__ C, unsigned short* __restrict__ Xn2) {
    int i = blockIdx.x;
    const float* row = C + (size_t)i * THREE_I;
    float vals[8], s1 = 0.0f, s2 = 0.0f;
    #pragma unroll
    for (int q = 0; q < 8; q++) {
        int c = threadIdx.x + q * 256;
        float a  = row[c];
        float bb = row[c + INTER];
        float cc = row[c + 2 * INTER];
        float r  = a > 0.0f ? a : 0.0f;
        float v  = r * r * r * bb + cc;
        vals[q] = v; s1 += v; s2 += v * v;
    }
    block_reduce2(s1, s2);
    float mean  = s1 * (1.0f / INTER);
    float var   = s2 - (float)INTER * mean * mean;
    float scale = sqrtf((float)INTER) * rsqrtf(var);
    #pragma unroll
    for (int q = 0; q < 8; q++)
        Xn2[(size_t)i * INTER + threadIdx.x + q * 256] = f2bf((vals[q] - mean) * scale);
}

// ---------------------------------------------------------------------------
// 5) causal grouped conv as 7 shifted GEMMs with a sliding A super-tile
// ---------------------------------------------------------------------------
__global__ __launch_bounds__(256) void conv_gemm_kernel(
        const unsigned short* __restrict__ Xn,   // [NTOK][INTER] bf16
        const unsigned short* __restrict__ W1s,  // [KW][THREE_I][GIC] bf16
        float* __restrict__ Out) {               // [NTOK][THREE_I] fp32
    __shared__ alignas(16) unsigned short As[MEXT * LDSK];
    __shared__ alignas(16) unsigned short Bs[NT * LDSK];

    const int tid  = threadIdx.x;
    const int lane = tid & 31;
    const int wave = tid >> 5;
    const int wm   = (wave & 1) * 64;
    const int wn   = (wave >> 1) * 32;

    const int n0    = blockIdx.x * NT;       // output channel tile (within a group)
    const int i0    = blockIdx.y * MT;       // token tile
    const int g     = n0 / GOC;              // conv group
    const int sbase = i0 & (SQ - 1);         // position within sequence

    v8f acc[4][2];
    #pragma unroll
    for (int i = 0; i < 4; i++)
        #pragma unroll
        for (int j = 0; j < 2; j++)
            #pragma unroll
            for (int v = 0; v < 8; v++) acc[i][j][v] = 0.0f;

    for (int ch = 0; ch < GIC; ch += KT) {
        __syncthreads();                     // previous readers of As/Bs done
        // A super-tile: rows r = 0..133 -> token i0-6+r (zero before sequence start)
        for (int p = 0; p < 5; p++) {
            int seg = tid + p * 256;
            if (seg < MEXT * 8) {
                int r = seg >> 3, s = seg & 7;
                if (sbase + r >= (KW - 1)) {
                    int tok = i0 - (KW - 1) + r;
                    cp16(Xn + (size_t)tok * INTER + g * GIC + ch + s * 8,
                         &As[r * LDSK + s * 8]);
                } else {
                    *(uint4*)&As[r * LDSK + s * 8] = make_uint4(0u, 0u, 0u, 0u);
                }
            }
        }
        for (int k = 0; k < KW; k++) {
            if (k > 0) __syncthreads();      // previous Bs readers done
            const unsigned short* Wk = W1s + (size_t)k * THREE_I * GIC;
            #pragma unroll
            for (int p = 0; p < 4; p++) {
                int seg = tid + p * 256;
                int r = seg >> 3, s = seg & 7;
                cp16(Wk + (size_t)(n0 + r) * GIC + ch + s * 8, &Bs[r * LDSK + s * 8]);
            }
            STAGE_WAIT_ALL();
            __syncthreads();
            gemm_tile_compute(As, Bs, k, wm, wn, lane, acc);
        }
    }

    const int half = lane >> 4, nl = lane & 15;
    #pragma unroll
    for (int i = 0; i < 4; i++)
        #pragma unroll
        for (int j = 0; j < 2; j++)
            #pragma unroll
            for (int v = 0; v < 8; v++) {
                int mrow = i0 + wm + i * 16 + v + 8 * half;
                Out[(size_t)mrow * THREE_I + n0 + wn + j * 16 + nl] = acc[i][j][v];
            }
}

// ---------------------------------------------------------------------------
// 6) final gather + transpose: out[b][c][s] = Z[perm[b*S+s]][c]
// ---------------------------------------------------------------------------
__global__ __launch_bounds__(256) void final_out_kernel(
        const float* __restrict__ Z, const int* __restrict__ perm,
        float* __restrict__ Out) {
    __shared__ float tile[32][33];
    int b = blockIdx.z, s0 = blockIdx.x * 32, c0 = blockIdx.y * 32;
    int tx = threadIdx.x, ty = threadIdx.y;
    #pragma unroll
    for (int yy = 0; yy < 4; yy++) {
        int s = s0 + ty + yy * 8;
        int src = perm[b * SQ + s];
        tile[ty + yy * 8][tx] = Z[(size_t)src * FEAT + c0 + tx];
    }
    __syncthreads();
    #pragma unroll
    for (int yy = 0; yy < 4; yy++) {
        int c = c0 + ty + yy * 8;
        Out[((size_t)b * FEAT + c) * SQ + s0 + tx] = tile[tx][ty + yy * 8];
    }
}

// ---------------------------------------------------------------------------
// Host launcher
// ---------------------------------------------------------------------------
extern "C" void kernel_launch(void* const* d_in, const int* in_sizes, int n_in,
                              void* d_out, int out_size, void* d_ws, size_t ws_size,
                              hipStream_t stream) {
    const float* inp = (const float*)d_in[0];
    const float* w0  = (const float*)d_in[1];
    const float* w1  = (const float*)d_in[2];
    const float* w2  = (const float*)d_in[3];
    const int*   sh0 = (const int*)d_in[4];
    const int*   sh2 = (const int*)d_in[5];
    const int*   prm = (const int*)d_in[6];
    float* out = (float*)d_out;

    char* ws = (char*)d_ws;
    // Region A (117,440,512 B): xT (16.8MB) + w0s (100.7MB); reused later by Z
    unsigned short* xT  = (unsigned short*)(ws + 0);
    unsigned short* w0s = (unsigned short*)(ws + 16777216);
    float*          z   = (float*)(ws + 0);
    // Region B (201,326,592 B): moe0 output; reused by conv output
    float* ofl   = (float*)(ws + 117440512);
    float* convo = ofl;
    // Region C (67,108,864 B): U; reused by xn2 (33.6MB) + w2s (33.6MB)
    float*          u   = (float*)(ws + 318767104);
    unsigned short* xn2 = (unsigned short*)(ws + 318767104);
    unsigned short* w2s = (unsigned short*)(ws + 318767104 + 33554432);
    // Region D: xn (33.6MB)
    unsigned short* xn  = (unsigned short*)(ws + 385875968);
    // Region E: w1s (44MB)
    unsigned short* w1s = (unsigned short*)(ws + 419430400);
    // Region F: inverse shuffles
    int* inv0 = (int*)(ws + 463470592);
    int* inv2 = inv0 + FEAT;

    // 0/1: prep
    inv_perm_kernel<<<dim3(8), dim3(256), 0, stream>>>(sh0, sh2, inv0, inv2);
    transpose_in_kernel<<<dim3(SQ / 32, FEAT / 32, BQ), dim3(32, 8), 0, stream>>>(inp, xT);
    repack_w0_kernel<<<dim3(THREE_I / 32, FEAT / 32, NEXP), dim3(32, 8), 0, stream>>>(w0, inv0, w0s);

    // 2: MoE #1  (M=1024/expert, K=1024, N=6144)
    moe_gemm_kernel<<<dim3(THREE_I / NT, (NTOK / NEXP) / MT, NEXP), dim3(256), 0, stream>>>(
        xT, w0s, prm, ofl, FEAT, THREE_I);

    // 3: cumsum gate + norm #1
    cumsum_gate_kernel<<<dim3(INTER / 256, BQ), dim3(256), 0, stream>>>(ofl, prm, u);
    norm1_kernel<<<dim3(NTOK), dim3(256), 0, stream>>>(u, xn);

    // 4: conv weights + conv (7 shifted GEMMs), overwrites region B
    repack_w1_kernel<<<dim3(THREE_I), dim3(128), 0, stream>>>(w1, w1s);
    conv_gemm_kernel<<<dim3(THREE_I / NT, NTOK / MT), dim3(256), 0, stream>>>(xn, w1s, convo);

    // 5: norm #2 (region C reuse) + w2 repack
    norm2_kernel<<<dim3(NTOK), dim3(256), 0, stream>>>(convo, xn2);
    repack_w2_kernel<<<dim3(FEAT / 32, INTER / 32, NEXP), dim3(32, 8), 0, stream>>>(w2, inv2, w2s);

    // 6: MoE #2  (M=1024/expert, K=2048, N=1024) into region A
    moe_gemm_kernel<<<dim3(FEAT / NT, (NTOK / NEXP) / MT, NEXP), dim3(256), 0, stream>>>(
        xn2, w2s, prm, z, INTER, FEAT);

    // 7: final gather + transpose to (B, FEAT, S)
    final_out_kernel<<<dim3(SQ / 32, FEAT / 32, BQ), dim3(32, 8), 0, stream>>>(z, prm, out);

    (void)in_sizes; (void)n_in; (void)out_size; (void)ws_size;
}
